// LSTMDecoderLuong_52991306498188
// MI455X (gfx1250) — compile-verified, matrix operand order checked by reading, not run on hardware
//
#include <hip/hip_runtime.h>

// ---------------- problem constants ----------------
static constexpr int Bc = 16;      // batch
static constexpr int Tc = 256;     // tgt seq len
static constexpr int Sc = 256;     // src seq len
static constexpr int Hc = 1024;    // hidden
static constexpr int Ec = 512;     // embed dim
static constexpr int G4H = 4096;   // 4*H
static constexpr int BT = Bc * Tc; // 4096

typedef __attribute__((ext_vector_type(16))) _Float16 v16h;
typedef __attribute__((ext_vector_type(8)))  _Float16 v8h;
typedef __attribute__((ext_vector_type(8)))  float    v8f;
typedef __attribute__((ext_vector_type(4)))  unsigned int u32x4;
typedef __attribute__((ext_vector_type(8)))  int i32x8;
typedef __attribute__((ext_vector_type(4)))  int i32x4;

#define WMMA_F16F32(A, B, C) \
  __builtin_amdgcn_wmma_f32_16x16x32_f16(false, (A), false, (B), (short)0, (C), false, false)

#if __has_builtin(__builtin_amdgcn_tensor_load_to_lds)
#define HAVE_TDM 1
#endif

// Load one 16x32 f16 fragment (A and B^T layouts are identical on CDNA5):
// lane 0-15 -> row +lane,   K: k+0..7 (VGPR0-3) and k+16..23 (VGPR4-7)
// lane16-31 -> row +lane-16, K: k+8..15 and k+24..31
static __device__ __forceinline__ v16h frag_load(const _Float16* base, int ld, int row, int k) {
  const int lane = threadIdx.x & 31;
  const int r  = row + (lane & 15);
  const int kk = k + ((lane >> 4) << 3);
  const _Float16* p = base + (size_t)r * ld + kk;
  union { v16h v; v8h h[2]; } u;
  u.h[0] = *reinterpret_cast<const v8h*>(p);        // b128 load (global or ds)
  u.h[1] = *reinterpret_cast<const v8h*>(p + 16);   // b128 load
  return u.v;
}

// Store one 16x16 f32 accumulator tile. D layout: lane holds col = lane&15,
// VGPR v holds row v (lanes 0-15) / v+8 (lanes 16-31).
template <bool OUT_F16, bool HAS_BIAS>
static __device__ __forceinline__ void store_tile(v8f acc, void* __restrict__ C,
                                                  int ldc, int m0, int ncol,
                                                  const float* __restrict__ bias) {
  const int lane = threadIdx.x & 31;
  const int col = ncol + (lane & 15);
  const int rb  = m0 + ((lane >> 4) << 3);
  const float bv = HAS_BIAS ? bias[col] : 0.0f;
#pragma unroll
  for (int v = 0; v < 8; ++v) {
    const float val = acc[v] + bv;
    if (OUT_F16) ((_Float16*)C)[(size_t)(rb + v) * ldc + col] = (_Float16)val;
    else         ((float*)C)[(size_t)(rb + v) * ldc + col] = val;
  }
}

// ---------------- generic (batched) WMMA GEMM ----------------
// C[M,N] = A[M,K] @ Bw[N,K]^T (+ bias[N]). 128 threads = 4 waves; each wave
// computes a 16x64 strip. Software-pipelined: k+32 fragments are loaded before
// the WMMAs on k's fragments so loads overlap matrix math (partial loadcnt
// waits instead of s_wait_loadcnt 0).
template <bool OUT_F16, bool HAS_BIAS>
__global__ __launch_bounds__(128) void gemm_wmma(
    const _Float16* __restrict__ A, int lda, size_t strideA,
    const _Float16* __restrict__ Bw, int ldb, size_t strideB,
    void* __restrict__ C, int ldc, size_t strideC,
    const float* __restrict__ bias, int M, int N, int K) {
  const int wave = threadIdx.x >> 5;
  const int n0 = (blockIdx.x * 4 + wave) * 64;
  const int m0 = blockIdx.y * 16;
  if (n0 >= N || m0 >= M) return;
  const int bz = blockIdx.z;
  A  += (size_t)bz * strideA;
  Bw += (size_t)bz * strideB;
  char* Cb = (char*)C + (size_t)bz * strideC * (OUT_F16 ? 2 : 4);

  v8f acc0 = {}, acc1 = {}, acc2 = {}, acc3 = {};
  v16h a  = frag_load(A, lda, m0, 0);
  v16h b0 = frag_load(Bw, ldb, n0 +  0, 0);
  v16h b1 = frag_load(Bw, ldb, n0 + 16, 0);
  v16h b2 = frag_load(Bw, ldb, n0 + 32, 0);
  v16h b3 = frag_load(Bw, ldb, n0 + 48, 0);
  for (int k = 32; k < K; k += 32) {
    v16h an  = frag_load(A, lda, m0, k);
    v16h bn0 = frag_load(Bw, ldb, n0 +  0, k);
    v16h bn1 = frag_load(Bw, ldb, n0 + 16, k);
    v16h bn2 = frag_load(Bw, ldb, n0 + 32, k);
    v16h bn3 = frag_load(Bw, ldb, n0 + 48, k);
    acc0 = WMMA_F16F32(a, b0, acc0);
    acc1 = WMMA_F16F32(a, b1, acc1);
    acc2 = WMMA_F16F32(a, b2, acc2);
    acc3 = WMMA_F16F32(a, b3, acc3);
    a = an; b0 = bn0; b1 = bn1; b2 = bn2; b3 = bn3;
  }
  acc0 = WMMA_F16F32(a, b0, acc0);
  acc1 = WMMA_F16F32(a, b1, acc1);
  acc2 = WMMA_F16F32(a, b2, acc2);
  acc3 = WMMA_F16F32(a, b3, acc3);

  store_tile<OUT_F16, HAS_BIAS>(acc0, Cb, ldc, m0, n0 +  0, bias);
  store_tile<OUT_F16, HAS_BIAS>(acc1, Cb, ldc, m0, n0 + 16, bias);
  store_tile<OUT_F16, HAS_BIAS>(acc2, Cb, ldc, m0, n0 + 32, bias);
  store_tile<OUT_F16, HAS_BIAS>(acc3, Cb, ldc, m0, n0 + 48, bias);
}

// ---------------- LSTM recurrent step ----------------
// h_prev (16x1024 f16 = 32KB) is staged once per block into LDS via the
// Tensor Data Mover (fallback: plain load+ds_store), then every wave's A
// fragments come from LDS (ds_load_b128) while W_hh B fragments stream from
// L2, software-pipelined against the WMMAs.
__global__ __launch_bounds__(128) void lstm_step(
    const float* __restrict__ xproj,        // [B*T, 4H], row = b*T + t
    const _Float16* __restrict__ Whh,       // [4H, H] f16
    const _Float16* __restrict__ h_prev,    // [16, H] f16 (read)
    _Float16* __restrict__ h_next,          // [16, H] f16 (write)
    float* __restrict__ c_state,            // [16, H] f32 (rmw, slice-owned)
    float* __restrict__ h_state,            // [16, H] f32 (write)
    _Float16* __restrict__ seq_out,         // f16 seq output, row (b*T+t)*ld
    int ld_seq, int t) {
  __shared__ __align__(16) _Float16 hsh[Bc * Hc];   // 32KB staged h_prev

#if defined(HAVE_TDM)
  if (threadIdx.x < 32) {                 // wave 0 drives the TDM
    typedef __attribute__((address_space(3))) void* as3_ptr;
    const unsigned lds_off =
        (unsigned)(unsigned long long)(as3_ptr)(void*)&hsh[0];
    const unsigned long long ga =
        ((unsigned long long)(uintptr_t)h_prev) & 0x1FFFFFFFFFFFFFFull;
    // D# group0: count=1 | lds_addr | global_addr[56:0] | type=2
    u32x4 g0;
    g0.x = 1u;
    g0.y = lds_off;
    g0.z = (unsigned)(ga & 0xFFFFFFFFull);
    g0.w = (unsigned)((ga >> 32) & 0x1FFFFFFull) | 0x80000000u;
    // D# group1: data_size=2B; 1 row of 16384 elems; tile = tensor
    i32x8 g1;
    g1[0] = 0x00010000;        // wg_mask=0, data_size=1 (2 bytes)
    g1[1] = 0x40000000;        // tensor_dim0[15:0]=16384 in bits[63:48]
    g1[2] = 0x00010000;        // tensor_dim1=1 in bits[95:80]
    g1[3] = 0x40000000;        // tile_dim0=16384 in bits[127:112]
    g1[4] = 1;                 // tile_dim1=1
    g1[5] = 16384;             // tensor_dim0_stride[31:0]
    g1[6] = 0;
    g1[7] = 0;
    const i32x4 z4 = {0, 0, 0, 0};
    const i32x8 z8 = {0, 0, 0, 0, 0, 0, 0, 0};
    // 6-arg (clang-23 / therock) form: groups 0-4 + cpol
    __builtin_amdgcn_tensor_load_to_lds(g0, g1, z4, z4, z8, 0);
#if __has_builtin(__builtin_amdgcn_s_wait_tensorcnt)
    __builtin_amdgcn_s_wait_tensorcnt(0);
#else
    asm volatile("s_wait_tensorcnt 0x0" ::: "memory");
#endif
  }
  __syncthreads();
#else
  for (int i = threadIdx.x; i < (Bc * Hc) / 8; i += blockDim.x)
    *reinterpret_cast<v8h*>(&hsh[i * 8]) =
        *reinterpret_cast<const v8h*>(&h_prev[i * 8]);
  __syncthreads();
#endif

  const int wave = threadIdx.x >> 5;
  const int tile = blockIdx.x * 4 + wave;   // 0..63 hidden-16 slice
  const int n0 = tile * 16;

  v8f acc0 = {}, acc1 = {}, acc2 = {}, acc3 = {};
  v16h a  = frag_load(hsh, Hc, 0, 0);                 // ds_load_b128 x2
  v16h b0 = frag_load(Whh, Hc, 0 * Hc + n0, 0);
  v16h b1 = frag_load(Whh, Hc, 1 * Hc + n0, 0);
  v16h b2 = frag_load(Whh, Hc, 2 * Hc + n0, 0);
  v16h b3 = frag_load(Whh, Hc, 3 * Hc + n0, 0);
  for (int k = 32; k < Hc; k += 32) {
    v16h an  = frag_load(hsh, Hc, 0, k);
    v16h bn0 = frag_load(Whh, Hc, 0 * Hc + n0, k);
    v16h bn1 = frag_load(Whh, Hc, 1 * Hc + n0, k);
    v16h bn2 = frag_load(Whh, Hc, 2 * Hc + n0, k);
    v16h bn3 = frag_load(Whh, Hc, 3 * Hc + n0, k);
    acc0 = WMMA_F16F32(a, b0, acc0);
    acc1 = WMMA_F16F32(a, b1, acc1);
    acc2 = WMMA_F16F32(a, b2, acc2);
    acc3 = WMMA_F16F32(a, b3, acc3);
    a = an; b0 = bn0; b1 = bn1; b2 = bn2; b3 = bn3;
  }
  acc0 = WMMA_F16F32(a, b0, acc0);
  acc1 = WMMA_F16F32(a, b1, acc1);
  acc2 = WMMA_F16F32(a, b2, acc2);
  acc3 = WMMA_F16F32(a, b3, acc3);

  const int lane = threadIdx.x & 31;
  const int col = n0 + (lane & 15);
  const int rb  = (lane >> 4) << 3;
#pragma unroll
  for (int v = 0; v < 8; ++v) {
    const int m = rb + v;                               // batch index 0..15
    const size_t xrow = ((size_t)m * Tc + t) * G4H;
    const float gi = xproj[xrow + 0 * Hc + col] + acc0[v];
    const float gf = xproj[xrow + 1 * Hc + col] + acc1[v];
    const float gg = xproj[xrow + 2 * Hc + col] + acc2[v];
    const float go = xproj[xrow + 3 * Hc + col] + acc3[v];
    const float i_ = 1.0f / (1.0f + __expf(-gi));
    const float f_ = 1.0f / (1.0f + __expf(-gf));
    const float o_ = 1.0f / (1.0f + __expf(-go));
    const float g_ = tanhf(gg);
    const size_t idx = (size_t)m * Hc + col;
    const float c = f_ * c_state[idx] + i_ * g_;
    c_state[idx] = c;
    const float h = o_ * tanhf(c);
    h_state[idx] = h;
    h_next[idx] = (_Float16)h;
    seq_out[((size_t)m * Tc + t) * ld_seq + col] = (_Float16)h;
  }
}

// ---------------- small helper kernels ----------------
__global__ void cvt_f32_f16(const float* __restrict__ in, _Float16* __restrict__ out, int n) {
  for (int i = blockIdx.x * blockDim.x + threadIdx.x; i < n; i += gridDim.x * blockDim.x)
    out[i] = (_Float16)in[i];
}

// dst[c*R + r] = src[r*C + c], batched over z
__global__ void transpose_cvt(const float* __restrict__ src, _Float16* __restrict__ dst,
                              int R, int C) {
  const size_t base = (size_t)blockIdx.z * R * C;
  for (int i = blockIdx.x * blockDim.x + threadIdx.x; i < R * C; i += gridDim.x * blockDim.x) {
    const int r = i / C, c = i % C;
    dst[base + (size_t)c * R + r] = (_Float16)src[base + i];
  }
}

__global__ void add_bias_vec(const float* __restrict__ a, const float* __restrict__ b,
                             float* __restrict__ o, int n) {
  const int i = blockIdx.x * blockDim.x + threadIdx.x;
  if (i < n) o[i] = a[i] + b[i];
}

__global__ void gather_embed(const int* __restrict__ tgt, const float* __restrict__ emb,
                             _Float16* __restrict__ out) {
  const int row = blockIdx.x;                 // 0..B*T-1
  const int tok = tgt[row];
  for (int e = threadIdx.x; e < Ec; e += blockDim.x)
    out[(size_t)row * Ec + e] = (_Float16)emb[(size_t)tok * Ec + e];
}

__global__ __launch_bounds__(256) void softmax_mask(
    const float* __restrict__ scores, const unsigned char* __restrict__ mask,
    _Float16* __restrict__ attn) {
  const int row = blockIdx.x;                 // b*T + t
  const int b = row / Tc;
  const int s = threadIdx.x;                  // S == 256 == blockDim
  float v = scores[(size_t)row * Sc + s];
  if (!mask[b * Sc + s]) v = -3.0e38f;
  __shared__ float red[Sc];
  red[s] = v; __syncthreads();
  for (int off = 128; off > 0; off >>= 1) {
    if (s < off) red[s] = fmaxf(red[s], red[s + off]);
    __syncthreads();
  }
  const float mx = red[0]; __syncthreads();
  const float e = __expf(v - mx);
  red[s] = e; __syncthreads();
  for (int off = 128; off > 0; off >>= 1) {
    if (s < off) red[s] += red[s + off];
    __syncthreads();
  }
  attn[(size_t)row * Sc + s] = (_Float16)(e / red[0]);
}

// ---------------- host launch ----------------
extern "C" void kernel_launch(void* const* d_in, const int* in_sizes, int n_in,
                              void* d_out, int out_size, void* d_ws, size_t ws_size,
                              hipStream_t stream) {
  const int*   tgt   = (const int*)d_in[0];
  const float* enc   = (const float*)d_in[1];
  const unsigned char* mask = (const unsigned char*)d_in[2];
  const float* emb   = (const float*)d_in[3];
  const float* Wih0  = (const float*)d_in[4];
  const float* Whh0  = (const float*)d_in[5];
  const float* bih0  = (const float*)d_in[6];
  const float* bhh0  = (const float*)d_in[7];
  const float* Wih1  = (const float*)d_in[8];
  const float* Whh1  = (const float*)d_in[9];
  const float* bih1  = (const float*)d_in[10];
  const float* bhh1  = (const float*)d_in[11];
  const float* Wattn = (const float*)d_in[12];
  const float* Wcat  = (const float*)d_in[13];
  const float* bcat  = (const float*)d_in[14];

  char* p = (char*)d_ws;
  auto alloc = [&](size_t bytes) { char* r = p; p += (bytes + 255) & ~(size_t)255; return r; };

  _Float16* embx   = (_Float16*)alloc((size_t)BT * Ec * 2);
  _Float16* wih0h  = (_Float16*)alloc((size_t)G4H * Ec * 2);
  _Float16* whh0h  = (_Float16*)alloc((size_t)G4H * Hc * 2);
  _Float16* wih1h  = (_Float16*)alloc((size_t)G4H * Hc * 2);
  _Float16* whh1h  = (_Float16*)alloc((size_t)G4H * Hc * 2);
  _Float16* wattnT = (_Float16*)alloc((size_t)Hc * Hc * 2);
  _Float16* wcath  = (_Float16*)alloc((size_t)Hc * 2 * Hc * 2);
  _Float16* ench   = (_Float16*)alloc((size_t)Bc * Sc * Hc * 2);
  _Float16* encT   = (_Float16*)alloc((size_t)Bc * Hc * Sc * 2);
  float*    bias0  = (float*)alloc(G4H * 4);
  float*    bias1  = (float*)alloc(G4H * 4);
  float*    xproj  = (float*)alloc((size_t)BT * G4H * 4);   // reused layer0/layer1
  _Float16* h0seq  = (_Float16*)alloc((size_t)BT * Hc * 2);
  _Float16* concat = (_Float16*)alloc((size_t)BT * 2 * Hc * 2); // [lstm_out | context]
  _Float16* energy = (_Float16*)alloc((size_t)BT * Hc * 2);
  float*    scores = (float*)alloc((size_t)BT * Sc * 4);
  _Float16* attn   = (_Float16*)alloc((size_t)BT * Sc * 2);
  _Float16* h16_0a = (_Float16*)alloc((size_t)Bc * Hc * 2);
  _Float16* h16_0b = (_Float16*)alloc((size_t)Bc * Hc * 2);
  _Float16* h16_1a = (_Float16*)alloc((size_t)Bc * Hc * 2);
  _Float16* h16_1b = (_Float16*)alloc((size_t)Bc * Hc * 2);
  float*    c0     = (float*)alloc((size_t)Bc * Hc * 4);
  float*    c1     = (float*)alloc((size_t)Bc * Hc * 4);
  float*    h0f    = (float*)alloc((size_t)Bc * Hc * 4);
  float*    h1f    = (float*)alloc((size_t)Bc * Hc * 4);

  // zero recurrent state (every call; deterministic)
  (void)hipMemsetAsync(h16_0a, 0, (size_t)Bc * Hc * 2, stream);
  (void)hipMemsetAsync(h16_0b, 0, (size_t)Bc * Hc * 2, stream);
  (void)hipMemsetAsync(h16_1a, 0, (size_t)Bc * Hc * 2, stream);
  (void)hipMemsetAsync(h16_1b, 0, (size_t)Bc * Hc * 2, stream);
  (void)hipMemsetAsync(c0, 0, (size_t)Bc * Hc * 4, stream);
  (void)hipMemsetAsync(c1, 0, (size_t)Bc * Hc * 4, stream);

  // weight / activation conversions
  auto cvt = [&](const float* s, _Float16* d, int n) {
    cvt_f32_f16<<<(n + 255) / 256, 256, 0, stream>>>(s, d, n);
  };
  cvt(Wih0, wih0h, G4H * Ec);
  cvt(Whh0, whh0h, G4H * Hc);
  cvt(Wih1, wih1h, G4H * Hc);
  cvt(Whh1, whh1h, G4H * Hc);
  cvt(Wcat, wcath, Hc * 2 * Hc);
  cvt(enc,  ench,  Bc * Sc * Hc);
  transpose_cvt<<<dim3(1024, 1, Bc), 256, 0, stream>>>(enc, encT, Sc, Hc);   // [S,H]->[H,S] per b
  transpose_cvt<<<dim3(4096, 1, 1), 256, 0, stream>>>(Wattn, wattnT, Hc, Hc);

  add_bias_vec<<<G4H / 256, 256, 0, stream>>>(bih0, bhh0, bias0, G4H);
  add_bias_vec<<<G4H / 256, 256, 0, stream>>>(bih1, bhh1, bias1, G4H);

  gather_embed<<<BT, 128, 0, stream>>>(tgt, emb, embx);

  // layer 0 input projection: [4096,512] @ [4096,512]^T + bias -> f32 [4096,4096]
  gemm_wmma<false, true><<<dim3(G4H / 256, BT / 16, 1), 128, 0, stream>>>(
      embx, Ec, 0, wih0h, Ec, 0, xproj, G4H, 0, bias0, BT, G4H, Ec);

  // layer 0 recurrence
  for (int t = 0; t < Tc; ++t) {
    const _Float16* hp = (t & 1) ? h16_0b : h16_0a;
    _Float16*       hn = (t & 1) ? h16_0a : h16_0b;
    lstm_step<<<16, 128, 0, stream>>>(xproj, whh0h, hp, hn, c0, h0f, h0seq, Hc, t);
  }

  // layer 1 input projection (xproj buffer reused): [4096,1024] @ [4096,1024]^T
  gemm_wmma<false, true><<<dim3(G4H / 256, BT / 16, 1), 128, 0, stream>>>(
      h0seq, Hc, 0, wih1h, Hc, 0, xproj, G4H, 0, bias1, BT, G4H, Hc);

  // layer 1 recurrence; h written into concat[:, 0:H] (ld = 2H)
  for (int t = 0; t < Tc; ++t) {
    const _Float16* hp = (t & 1) ? h16_1b : h16_1a;
    _Float16*       hn = (t & 1) ? h16_1a : h16_1b;
    lstm_step<<<16, 128, 0, stream>>>(xproj, whh1h, hp, hn, c1, h1f, concat, 2 * Hc, t);
  }

  // final states -> d_out (dec_out | h0 h1 | c0 c1)
  float* outf = (float*)d_out;
  float* hfin = outf + (size_t)BT * Hc;
  (void)hipMemcpyAsync(hfin,               h0f, (size_t)Bc * Hc * 4, hipMemcpyDeviceToDevice, stream);
  (void)hipMemcpyAsync(hfin + Bc * Hc,     h1f, (size_t)Bc * Hc * 4, hipMemcpyDeviceToDevice, stream);
  (void)hipMemcpyAsync(hfin + 2 * Bc * Hc, c0,  (size_t)Bc * Hc * 4, hipMemcpyDeviceToDevice, stream);
  (void)hipMemcpyAsync(hfin + 3 * Bc * Hc, c1,  (size_t)Bc * Hc * 4, hipMemcpyDeviceToDevice, stream);

  // energy = lstm_out @ W_attn  (A from concat, lda=2H; W_attn pre-transposed)
  gemm_wmma<true, false><<<dim3(Hc / 256, BT / 16, 1), 128, 0, stream>>>(
      concat, 2 * Hc, 0, wattnT, Hc, 0, energy, Hc, 0, nullptr, BT, Hc, Hc);

  // scores[b] = energy[b] @ enc[b]^T : batched, M=T, N=S, K=H -> f32
  gemm_wmma<false, false><<<dim3(Sc / 256, Tc / 16, Bc), 128, 0, stream>>>(
      energy, Hc, (size_t)Tc * Hc, ench, Hc, (size_t)Sc * Hc,
      scores, Sc, (size_t)Tc * Sc, nullptr, Tc, Sc, Hc);

  softmax_mask<<<BT, Sc, 0, stream>>>(scores, mask, attn);

  // context[b] = attn[b] @ encT[b]^T : M=T, N=H, K=S -> f16 into concat[:, H:2H]
  gemm_wmma<true, false><<<dim3(Hc / 256, Tc / 16, Bc), 128, 0, stream>>>(
      attn, Sc, (size_t)Tc * Sc, encT, Sc, (size_t)Hc * Sc,
      concat + Hc, 2 * Hc, (size_t)Tc * 2 * Hc, nullptr, Tc, Hc, Sc);

  // decoder_outputs = concat @ W_concat^T + b_concat -> f32 d_out
  gemm_wmma<false, true><<<dim3(Hc / 256, BT / 16, 1), 128, 0, stream>>>(
      concat, 2 * Hc, 0, wcath, 2 * Hc, 0, outf, Hc, 0, bcat, BT, Hc, 2 * Hc);
}
